// GMAN_12833362280587
// MI455X (gfx1250) — compile-verified, matrix operand order checked by reading, not run on hardware
//
#include <hip/hip_runtime.h>
#include <hip/hip_bf16.h>
#include <math.h>
#include <stdint.h>

typedef __attribute__((ext_vector_type(16))) _Float16 v16h;
typedef __attribute__((ext_vector_type(8)))  float    v8f;

#define NHEAD   8
#define DHEAD   32
#define T_STEPS 12
#define N_NODES 512
#define B_BATCH 4
#define MROWS   (B_BATCH * T_STEPS * N_NODES)   // 24576

// K-packing for 16-bit WMMA fragments (ISA 7.12.2, 16-bit A-matrix 16x32):
// lanes 0-15: v0:K0,1 v1:K2,3 v2:K4,5 v3:K6,7 v4:K16,17 ... v7:K22,23
// lanes 16-31: same + 8
static __device__ __forceinline__ int kpack_base(int j, int hf) {
  return ((j < 4) ? (2 * j) : (16 + 2 * (j - 4))) + 8 * hf;
}

// gfx1250 async copy global->LDS, 16B per lane, tracked by ASYNCcnt.
// LDS address = low 32 bits of the generic pointer (ISA 10.2: LDS aperture
// addresses truncate to the LDS offset).
static __device__ __forceinline__ void async_b128(const _Float16* sbase,
                                                  uint32_t voff_bytes,
                                                  void* lds_ptr) {
  uint32_t lds = (uint32_t)(uintptr_t)lds_ptr;
  asm volatile("global_load_async_to_lds_b128 %0, %1, %2"
               :
               : "v"(lds), "v"(voff_bytes), "s"(sbase)
               : "memory");
}
static __device__ __forceinline__ void wait_async0() {
  asm volatile("s_wait_asynccnt 0" ::: "memory");
}
static __device__ __forceinline__ void wait_async4() {
  asm volatile("s_wait_asynccnt 4" ::: "memory");
}

// ---------------------------------------------------------------------------
// Weight convert: W (Kd x N=256, fp32 row-major) -> Wt (N x Kd, f16) transposed
// ---------------------------------------------------------------------------
__global__ void wconv_kernel(const float* __restrict__ W, _Float16* __restrict__ Wt,
                             int Kd, int N) {
  int i = blockIdx.x * 256 + threadIdx.x;
  if (i < Kd * N) {
    int n = i / Kd, k = i - n * Kd;
    Wt[i] = (_Float16)W[(long)k * N + n];
  }
}

__global__ void f32to16_kernel(const float* __restrict__ src,
                               _Float16* __restrict__ dst, int n) {
  int i = blockIdx.x * 256 + threadIdx.x;
  if (i < n) dst[i] = (_Float16)src[i];
}

// ---------------------------------------------------------------------------
// GEMM: C[M,N=256] (f16) = act( [A0|A1] @ W + bias (+ C if beta) )
// A f16, W f16 (pre-transposed Wt[n][k]), fp32 accumulation.
// Block tile 128x128, 8 waves x (2x4) wmma tiles, K-step 32.
// LDS double-buffered, filled with global_load_async_to_lds_b128.
// act: 0=none 1=relu 2=sigmoid.
// ---------------------------------------------------------------------------
__global__ __launch_bounds__(256)
void gemm_f16(const _Float16* __restrict__ A0, const _Float16* __restrict__ A1,
              int K0, int Kd, const _Float16* __restrict__ Wt,
              const float* __restrict__ bias, _Float16* __restrict__ C,
              int M, int N, int act, int beta) {
  __shared__ _Float16 As[2][128][40];   // 40-halfs stride: 16B aligned rows
  __shared__ _Float16 Ws[2][128][40];
  const int tid  = threadIdx.x;
  const int lane = tid & 31;
  const int wave = tid >> 5;
  const int hf = lane >> 4;
  const int ln = lane & 15;
  const int m0 = blockIdx.x * 128;
  const int n0 = blockIdx.y * 128;
  const int wr = (wave & 3) * 32;    // wave row offset (4 groups x 32)
  const int wc = (wave >> 2) * 64;   // wave col offset (2 groups x 64)
  const int K1 = Kd - K0;

  auto stage = [&](int buf, int kk) {
    const _Float16* Ab = (kk < K0) ? A0 : A1;
    const int kloc = (kk < K0) ? kk : kk - K0;
    const int w    = (kk < K0) ? K0 : K1;
    // A tile: 128 rows x 32 halfs = 512 chunks of 16B (2 per thread)
    for (int c = tid; c < 512; c += 256) {
      int r = c >> 2, c4 = c & 3;
      async_b128(Ab, (uint32_t)(((m0 + r) * w + kloc + c4 * 8) * 2),
                 &As[buf][r][c4 * 8]);
    }
    // W tile: 128 cols x 32 halfs
    for (int c = tid; c < 512; c += 256) {
      int r = c >> 2, c4 = c & 3;
      async_b128(Wt, (uint32_t)(((n0 + r) * Kd + kk + c4 * 8) * 2),
                 &Ws[buf][r][c4 * 8]);
    }
  };

  v8f zero = {};
  v8f acc[2][4];
#pragma unroll
  for (int a = 0; a < 2; ++a)
#pragma unroll
    for (int b = 0; b < 4; ++b) acc[a][b] = zero;

  const int ntile = Kd >> 5;
  stage(0, 0);
  for (int it = 0; it < ntile; ++it) {
    const int buf = it & 1;
    if (it + 1 < ntile) {
      stage(buf ^ 1, (it + 1) << 5);  // prefetch next tile into other buffer
      wait_async4();                  // previous tile complete, next in flight
    } else {
      wait_async0();
    }
    __syncthreads();

    v16h afr[2], bfr[4];
#pragma unroll
    for (int t = 0; t < 2; ++t) {
      int row = wr + t * 16 + ln;
#pragma unroll
      for (int j = 0; j < 8; ++j) {
        int kb = kpack_base(j, hf);
        afr[t][2 * j]     = As[buf][row][kb];
        afr[t][2 * j + 1] = As[buf][row][kb + 1];
      }
    }
#pragma unroll
    for (int u = 0; u < 4; ++u) {
      int col = wc + u * 16 + ln;
#pragma unroll
      for (int j = 0; j < 8; ++j) {
        int kb = kpack_base(j, hf);
        bfr[u][2 * j]     = Ws[buf][col][kb];
        bfr[u][2 * j + 1] = Ws[buf][col][kb + 1];
      }
    }
#pragma unroll
    for (int a = 0; a < 2; ++a)
#pragma unroll
      for (int b = 0; b < 4; ++b)
        acc[a][b] = __builtin_amdgcn_wmma_f32_16x16x32_f16(
            false, afr[a], false, bfr[b], (short)0, acc[a][b], false, false);
    __syncthreads();
  }

#pragma unroll
  for (int a = 0; a < 2; ++a) {
#pragma unroll
    for (int b = 0; b < 4; ++b) {
#pragma unroll
      for (int i = 0; i < 8; ++i) {
        int row = m0 + wr + a * 16 + i + 8 * hf;  // C layout: vgpr i -> M=i (+8 hi)
        int col = n0 + wc + b * 16 + ln;
        if (row < M) {
          long idx = (long)row * N + col;
          float v = acc[a][b][i] + bias[col];
          if (beta) v += (float)C[idx];
          if (act == 1)      v = fmaxf(v, 0.f);
          else if (act == 2) v = 1.f / (1.f + __expf(-v));
          C[idx] = (_Float16)v;
        }
      }
    }
  }
}

// ---------------------------------------------------------------------------
// Flash spatial attention: per (b,t,head): softmax(Q K^T / sqrt(32)) V
// Q,K,V,O f16, layout (B,T,N,256), head h at cols [32h,32h+32).
// grid = (4 qtiles, 8 heads, 48 bt), 256 threads (8 waves x 16 query rows).
// Q/K/V staged with async-to-LDS; QK^T and P.V through wmma f16, fp32 stats.
// ---------------------------------------------------------------------------
__global__ __launch_bounds__(256)
void flash_spatial(const _Float16* __restrict__ Q, const _Float16* __restrict__ K,
                   const _Float16* __restrict__ V, _Float16* __restrict__ O) {
  __shared__ _Float16 Qs[128][40];
  __shared__ _Float16 Ks[64][40];
  __shared__ _Float16 Vs[64][40];
  __shared__ float    Ss[128][66];
  __shared__ _Float16 Ps[128][66];
  __shared__ float mrow[128], lrow[128], arow[128];

  const int tid  = threadIdx.x;
  const int lane = tid & 31;
  const int wave = tid >> 5;
  const int hf = lane >> 4;
  const int ln = lane & 15;
  const int qt = blockIdx.x;
  const int h  = blockIdx.y;
  const int bt = blockIdx.z;
  const long base = ((long)bt * N_NODES) << 8;
  const int  q0   = qt * 128;
  const float scale = 0.17677669529663687f;  // 1/sqrt(32)

  const _Float16* Qbase = Q + base + h * DHEAD;
  const _Float16* Kbase = K + base + h * DHEAD;
  const _Float16* Vbase = V + base + h * DHEAD;

  // async-stage Q tile (128x32 halfs = 512 x16B chunks, 2 per thread)
  for (int c = tid; c < 512; c += 256) {
    int r = c >> 2, c4 = c & 3;
    async_b128(Qbase, (uint32_t)((((q0 + r) << 8) + c4 * 8) * 2), &Qs[r][c4 * 8]);
  }
  if (tid < 128) { mrow[tid] = -1e30f; lrow[tid] = 0.f; }

  const int r0 = wave * 16;
  v16h qf;
  v8f zero = {};
  v8f oacc[2];
  oacc[0] = zero; oacc[1] = zero;

  for (int kb0 = 0; kb0 < N_NODES; kb0 += 64) {
    // async-stage K/V block (64x32 halfs = 256 chunks, 1 per thread each)
    {
      int r = tid >> 2, c4 = tid & 3;
      uint32_t voff = (uint32_t)((((kb0 + r) << 8) + c4 * 8) * 2);
      async_b128(Kbase, voff, &Ks[r][c4 * 8]);
      async_b128(Vbase, voff, &Vs[r][c4 * 8]);
    }
    wait_async0();
    __syncthreads();

    if (kb0 == 0) {  // Q now resident: build per-wave A fragment once
#pragma unroll
      for (int j = 0; j < 8; ++j) {
        int kb = kpack_base(j, hf);
        qf[2 * j]     = Qs[r0 + ln][kb];
        qf[2 * j + 1] = Qs[r0 + ln][kb + 1];
      }
    }

    // S = Q K^T : B-fragment col n of K^T == row n of K (same packing as A)
#pragma unroll
    for (int ct = 0; ct < 4; ++ct) {
      v16h bf;
#pragma unroll
      for (int j = 0; j < 8; ++j) {
        int kb = kpack_base(j, hf);
        bf[2 * j]     = Ks[ct * 16 + ln][kb];
        bf[2 * j + 1] = Ks[ct * 16 + ln][kb + 1];
      }
      v8f sc = zero;
      sc = __builtin_amdgcn_wmma_f32_16x16x32_f16(false, qf, false, bf,
                                                  (short)0, sc, false, false);
#pragma unroll
      for (int i = 0; i < 8; ++i)
        Ss[r0 + i + 8 * hf][ct * 16 + ln] = sc[i] * scale;
    }
    __syncthreads();

    // online softmax (one thread per query row)
    if (tid < 128) {
      float mo = mrow[tid], bm = -1e30f;
      for (int c = 0; c < 64; ++c) bm = fmaxf(bm, Ss[tid][c]);
      float mn = fmaxf(mo, bm);
      float al = __expf(mo - mn);
      float sum = 0.f;
      for (int c = 0; c < 64; ++c) {
        float p = __expf(Ss[tid][c] - mn);
        Ps[tid][c] = (_Float16)p;
        sum += p;
      }
      mrow[tid] = mn;
      lrow[tid] = lrow[tid] * al + sum;
      arow[tid] = al;
    }
    __syncthreads();

    // O = O*alpha + P V   (16 rows per wave, N=32, K'=64)
    float alv[8];
#pragma unroll
    for (int i = 0; i < 8; ++i) alv[i] = arow[r0 + i + 8 * hf];
#pragma unroll
    for (int c = 0; c < 2; ++c)
#pragma unroll
      for (int i = 0; i < 8; ++i) oacc[c][i] *= alv[i];
#pragma unroll
    for (int ks = 0; ks < 2; ++ks) {
      v16h pf;
#pragma unroll
      for (int j = 0; j < 8; ++j) {
        int kb = kpack_base(j, hf);
        pf[2 * j]     = Ps[r0 + ln][ks * 32 + kb];
        pf[2 * j + 1] = Ps[r0 + ln][ks * 32 + kb + 1];
      }
#pragma unroll
      for (int c = 0; c < 2; ++c) {
        v16h vf;
#pragma unroll
        for (int j = 0; j < 8; ++j) {
          int kb = kpack_base(j, hf);
          vf[2 * j]     = Vs[ks * 32 + kb][c * 16 + ln];
          vf[2 * j + 1] = Vs[ks * 32 + kb + 1][c * 16 + ln];
        }
        oacc[c] = __builtin_amdgcn_wmma_f32_16x16x32_f16(
            false, pf, false, vf, (short)0, oacc[c], false, false);
      }
    }
    __syncthreads();
  }

#pragma unroll
  for (int c = 0; c < 2; ++c)
#pragma unroll
    for (int i = 0; i < 8; ++i) {
      int r = r0 + i + 8 * hf;
      O[base + (((long)(q0 + r)) << 8) + h * DHEAD + c * 16 + ln] =
          (_Float16)(oacc[c][i] / lrow[r]);
    }
}

// ---------------------------------------------------------------------------
// Tiny attentions over the 12 time steps (temporal + transform):
// one thread per (b, n, head, t_query).  ~0.5% of total FLOPs.
// ---------------------------------------------------------------------------
__global__ __launch_bounds__(128)
void small_attn(const _Float16* __restrict__ Q, const _Float16* __restrict__ K,
                const _Float16* __restrict__ V, _Float16* __restrict__ O) {
  int gid = blockIdx.x * 128 + threadIdx.x;
  if (gid >= B_BATCH * N_NODES * NHEAD * T_STEPS) return;
  int t = gid % T_STEPS; int tmp = gid / T_STEPS;
  int h = tmp % NHEAD;   tmp /= NHEAD;
  int n = tmp % N_NODES; int b = tmp / N_NODES;
  const float scale = 0.17677669529663687f;
  long qoff = (((long)(b * T_STEPS + t) * N_NODES + n) << 8) + h * DHEAD;
  float q[DHEAD];
#pragma unroll
  for (int d = 0; d < DHEAD; ++d) q[d] = (float)Q[qoff + d];
  float s[T_STEPS]; float m = -1e30f;
  for (int u = 0; u < T_STEPS; ++u) {
    long koff = (((long)(b * T_STEPS + u) * N_NODES + n) << 8) + h * DHEAD;
    float a = 0.f;
#pragma unroll
    for (int d = 0; d < DHEAD; ++d) a += q[d] * (float)K[koff + d];
    a *= scale; s[u] = a; m = fmaxf(m, a);
  }
  float l = 0.f;
  for (int u = 0; u < T_STEPS; ++u) { s[u] = __expf(s[u] - m); l += s[u]; }
  float inv = 1.f / l;
  float o[DHEAD];
#pragma unroll
  for (int d = 0; d < DHEAD; ++d) o[d] = 0.f;
  for (int u = 0; u < T_STEPS; ++u) {
    long voff = (((long)(b * T_STEPS + u) * N_NODES + n) << 8) + h * DHEAD;
    float p = s[u] * inv;
#pragma unroll
    for (int d = 0; d < DHEAD; ++d) o[d] += p * (float)V[voff + d];
  }
#pragma unroll
  for (int d = 0; d < DHEAD; ++d) O[qoff + d] = (_Float16)o[d];
}

// ------------------------------- glue kernels ------------------------------
__global__ void fc1_kernel(const float* __restrict__ X, const float* __restrict__ w,
                           const float* __restrict__ b, _Float16* __restrict__ out,
                           int M) {
  long i = (long)blockIdx.x * 256 + threadIdx.x;
  if (i < (long)M * 256) {
    int r = (int)(i >> 8), j = (int)(i & 255);
    out[i] = (_Float16)fmaxf(X[r] * w[j] + b[j], 0.f);
  }
}

__global__ void te_kernel(const int* __restrict__ TE, const float* __restrict__ W5,
                          const float* __restrict__ b5, _Float16* __restrict__ te1) {
  int i = blockIdx.x * 256 + threadIdx.x;  // 96*256 rows (b, 2T)
  if (i < 96 * 256) {
    int r = i >> 8, j = i & 255;
    int dow = TE[r * 2 + 0], tod = TE[r * 2 + 1];
    te1[i] = (_Float16)fmaxf(W5[dow * 256 + j] + W5[(7 + tod) * 256 + j] + b5[j], 0.f);
  }
}

__global__ void ste_add_kernel(const _Float16* __restrict__ se2,
                               const _Float16* __restrict__ te2,
                               _Float16* __restrict__ STEP,
                               _Float16* __restrict__ STEQ) {
  long i = (long)blockIdx.x * 256 + threadIdx.x;
  if (i < (long)MROWS * 256) {
    int d = (int)(i & 255);
    long r = i >> 8;
    int n = (int)(r % N_NODES);
    long bt = r / N_NODES;
    int t = (int)(bt % T_STEPS), b = (int)(bt / T_STEPS);
    float se = (float)se2[n * 256 + d];
    STEP[i] = (_Float16)(se + (float)te2[(b * 24 + t) * 256 + d]);
    STEQ[i] = (_Float16)(se + (float)te2[(b * 24 + 12 + t) * 256 + d]);
  }
}

__global__ void gated_mix_kernel(_Float16* __restrict__ Z,
                                 const _Float16* __restrict__ HS,
                                 const _Float16* __restrict__ HT, long n) {
  long i = (long)blockIdx.x * 256 + threadIdx.x;
  if (i < n) {
    float z = (float)Z[i];
    Z[i] = (_Float16)(z * (float)HS[i] + (1.f - z) * (float)HT[i]);
  }
}

__global__ void resid_add_kernel(_Float16* __restrict__ X,
                                 const _Float16* __restrict__ G, long n) {
  long i = (long)blockIdx.x * 256 + threadIdx.x;
  if (i < n) X[i] = (_Float16)((float)X[i] + (float)G[i]);
}

__global__ void fc27_kernel(const _Float16* __restrict__ T1, const float* __restrict__ w,
                            const float* __restrict__ b, float* __restrict__ out, int M) {
  int r = blockIdx.x * 256 + threadIdx.x;
  if (r < M) {
    float acc = b[0];
    const _Float16* row = T1 + (long)r * 256;
    for (int d = 0; d < 256; ++d) acc += (float)row[d] * w[d];
    out[r] = acc;
  }
}

// ---------------------------------------------------------------------------
extern "C" void kernel_launch(void* const* d_in, const int* in_sizes, int n_in,
                              void* d_out, int out_size, void* d_ws, size_t ws_size,
                              hipStream_t stream) {
  const float* X  = (const float*)d_in[0];
  const float* SE = (const float*)d_in[1];
  const int*   TE = (const int*)d_in[2];
  auto W = [&](int i) { return (const float*)d_in[i]; };

  // JAX pytree (sorted-key) flat order of params:
  // dec(gf:3..10, sp:11..20, tp:21..30), enc(gf:31..38, sp:39..48, tp:49..58),
  // fc1(59,60) fc2(61,62) fc26(63,64) fc27(65,66),
  // ste(fc3:67,68 fc4:69,70 fc5:71,72 fc6:73,74), trans(k,o1,o2,q,v: 75..84)
  // each linear contributes (b, w); att2 sorted order: k,o1,o2,q,v

  const long MD = (long)MROWS * 256;
  _Float16* P    = (_Float16*)d_ws;
  _Float16* xcur = P + 0 * MD;
  _Float16* STEP = P + 1 * MD;
  _Float16* STEQ = P + 2 * MD;
  _Float16* Qb   = P + 3 * MD;
  _Float16* Kb   = P + 4 * MD;
  _Float16* Vb   = P + 5 * MD;
  _Float16* Ob   = P + 6 * MD;
  _Float16* T1   = P + 7 * MD;
  _Float16* HS   = P + 8 * MD;
  _Float16* HT   = P + 9 * MD;
  _Float16* Zb   = P + 10 * MD;
  _Float16* se1  = P + 11 * MD;         // 512*256
  _Float16* se2  = se1 + 512 * 256;     // 512*256
  _Float16* te1  = se2 + 512 * 256;     // 96*256
  _Float16* te2  = te1 + 96 * 256;      // 96*256
  _Float16* se0  = te2 + 96 * 256;      // 512*64  (SE converted to f16)
  _Float16* wf16 = se0 + 512 * 64;      // up to 512*256 (per-layer weight)

  auto linear = [&](const _Float16* A0, const _Float16* A1, int K0, int Kd,
                    int bi, int wi, _Float16* C, int M, int act, int beta) {
    wconv_kernel<<<(Kd * 256 + 255) / 256, 256, 0, stream>>>(W(wi), wf16, Kd, 256);
    dim3 g((M + 127) / 128, 2);  // N=256 -> 2 col tiles of 128
    gemm_f16<<<g, 256, 0, stream>>>(A0, A1 ? A1 : A0, A1 ? K0 : Kd, Kd, wf16,
                                    W(bi), C, M, 256, act, beta);
  };

  const long nEl = MD;
  const int ebk  = (int)((nEl + 255) / 256);

  // x = fc2(relu(fc1(X)))
  fc1_kernel<<<ebk, 256, 0, stream>>>(X, W(60), W(59), T1, MROWS);
  linear(T1, nullptr, 0, 256, 61, 62, xcur, MROWS, 0, 0);

  // STE embedding
  f32to16_kernel<<<(512 * 64 + 255) / 256, 256, 0, stream>>>(SE, se0, 512 * 64);
  linear(se0, nullptr, 0, 64, 67, 68, se1, 512, 1, 0);
  linear(se1, nullptr, 0, 256, 69, 70, se2, 512, 0, 0);
  te_kernel<<<(96 * 256 + 255) / 256, 256, 0, stream>>>(TE, W(72), W(71), te1);
  linear(te1, nullptr, 0, 256, 73, 74, te2, 96, 0, 0);
  ste_add_kernel<<<ebk, 256, 0, stream>>>(se2, te2, STEP, STEQ);

  dim3 fgrid(4, 8, 48);
  const int sblk = (B_BATCH * N_NODES * NHEAD * T_STEPS + 127) / 128;

  auto statt = [&](int base, const _Float16* ste) {
    int gf = base, sp = base + 8, tp = base + 18;
    // spatial attention
    linear(xcur, ste, 256, 512, sp + 6, sp + 7, Qb, MROWS, 1, 0);  // q
    linear(xcur, ste, 256, 512, sp + 0, sp + 1, Kb, MROWS, 1, 0);  // k
    linear(xcur, ste, 256, 512, sp + 8, sp + 9, Vb, MROWS, 1, 0);  // v
    flash_spatial<<<fgrid, 256, 0, stream>>>(Qb, Kb, Vb, Ob);
    linear(Ob, nullptr, 0, 256, sp + 2, sp + 3, T1, MROWS, 1, 0);  // o1 + relu
    linear(T1, nullptr, 0, 256, sp + 4, sp + 5, HS, MROWS, 0, 0);  // o2
    // temporal attention
    linear(xcur, ste, 256, 512, tp + 6, tp + 7, Qb, MROWS, 1, 0);
    linear(xcur, ste, 256, 512, tp + 0, tp + 1, Kb, MROWS, 1, 0);
    linear(xcur, ste, 256, 512, tp + 8, tp + 9, Vb, MROWS, 1, 0);
    small_attn<<<sblk, 128, 0, stream>>>(Qb, Kb, Vb, Ob);
    linear(Ob, nullptr, 0, 256, tp + 2, tp + 3, T1, MROWS, 1, 0);
    linear(T1, nullptr, 0, 256, tp + 4, tp + 5, HT, MROWS, 0, 0);
    // gated fusion: z = sigmoid(fc17(HS)+fc18(HT)); H = z*HS+(1-z)*HT
    linear(HS, nullptr, 0, 256, gf + 0, gf + 1, Zb, MROWS, 0, 0);
    linear(HT, nullptr, 0, 256, gf + 2, gf + 3, Zb, MROWS, 2, 1);
    gated_mix_kernel<<<ebk, 256, 0, stream>>>(Zb, HS, HT, nEl);
    linear(Zb, nullptr, 0, 256, gf + 4, gf + 5, T1, MROWS, 1, 0);  // fc19+relu
    linear(T1, nullptr, 0, 256, gf + 6, gf + 7, Ob, MROWS, 0, 0);  // fc20
    resid_add_kernel<<<ebk, 256, 0, stream>>>(xcur, Ob, nEl);
  };

  // encoder
  statt(31, STEP);

  // transform attention
  linear(STEQ, nullptr, 0, 256, 81, 82, Qb, MROWS, 1, 0);
  linear(STEP, nullptr, 0, 256, 75, 76, Kb, MROWS, 1, 0);
  linear(xcur, nullptr, 0, 256, 83, 84, Vb, MROWS, 1, 0);
  small_attn<<<sblk, 128, 0, stream>>>(Qb, Kb, Vb, Ob);
  linear(Ob, nullptr, 0, 256, 77, 78, T1, MROWS, 1, 0);
  linear(T1, nullptr, 0, 256, 79, 80, xcur, MROWS, 0, 0);

  // decoder
  statt(3, STEQ);

  // head
  linear(xcur, nullptr, 0, 256, 63, 64, T1, MROWS, 1, 0);
  fc27_kernel<<<(MROWS + 255) / 256, 256, 0, stream>>>(T1, W(66), W(65),
                                                       (float*)d_out, MROWS);
}
